// APD2Net_82815559401762
// MI455X (gfx1250) — compile-verified, matrix operand order checked by reading, not run on hardware
//
#include <hip/hip_runtime.h>
#include <hip/hip_bf16.h>
#include <stdint.h>

// ---------------------------------------------------------------------------
// Types for CDNA5 WMMA (gfx1250, wave32)
// ---------------------------------------------------------------------------
typedef __attribute__((ext_vector_type(16))) __bf16 v16bf;
typedef __attribute__((ext_vector_type(8)))  float  v8f;

union Frag16 {
    v16bf    bf;
    uint32_t u[8];
};

// ---------------------------------------------------------------------------
// Optional gfx1250 async global->LDS staging (guarded; sync fallback)
// Builtin prototype (per hipcc diagnostic): (as1 int*, as3 int*, imm, imm)
// ---------------------------------------------------------------------------
#if defined(__has_builtin)
#if __has_builtin(__builtin_amdgcn_global_load_async_to_lds_b32)
#define HAVE_ASYNC_B32 1
#endif
#endif

typedef __attribute__((address_space(1))) int as1_int;
typedef __attribute__((address_space(3))) int as3_int;

#if defined(HAVE_ASYNC_B32)
__device__ __forceinline__ void async_copy_b32(const uint32_t* g, uint32_t* l) {
    __builtin_amdgcn_global_load_async_to_lds_b32(
        (as1_int*)(uintptr_t)g,
        (as3_int*)(uint32_t)(uintptr_t)l,
        0, 0);
}
__device__ __forceinline__ void wait_async_lds() {
#if __has_builtin(__builtin_amdgcn_s_wait_asynccnt)
    __builtin_amdgcn_s_wait_asynccnt(0);
#else
    asm volatile("s_wait_asynccnt 0x0" ::: "memory");
#endif
}
#endif

// ---------------------------------------------------------------------------
// bf16 helpers (round-to-nearest-even)
// ---------------------------------------------------------------------------
__device__ __forceinline__ uint16_t f32_to_bf16(float f) {
    uint32_t x = __float_as_uint(f);
    uint32_t r = x + 0x7FFFu + ((x >> 16) & 1u);
    return (uint16_t)(r >> 16);
}
__device__ __forceinline__ float bf16_to_f32(uint16_t h) {
    return __uint_as_float(((uint32_t)h) << 16);
}

// ---------------------------------------------------------------------------
// Weight repack: w (Cout, Cin, 9) f32 -> swizzled bf16 B-fragment words.
// Output layout: [kc][ntile(=Cout/16)][vgpr(8)][lane(32)] as uint32 (2 bf16).
// K dimension = tap*Cin + c (tap-major), zero-padded to a multiple of 32.
// B fragment (16-bit 32x16, ISA 7.12.2): lane&15 = column N, vgpr v holds the
// K pair base(v) = (v<4 ? 2v : 16+2(v-4)), +8 for lanes 16..31.
// ---------------------------------------------------------------------------
__global__ void repack_weights(const float* __restrict__ w,
                               uint32_t* __restrict__ wsw,
                               int Cin, int Cout, long total) {
    long idx = (long)blockIdx.x * 256 + threadIdx.x;
    if (idx >= total) return;
    int  Ktot = 9 * Cin;
    int  NT   = Cout >> 4;
    int  lane = (int)(idx & 31);
    int  v    = (int)((idx >> 5) & 7);
    long t    = idx >> 8;
    int  ntile = (int)(t % NT);
    int  kc    = (int)(t / NT);
    int  n  = ntile * 16 + (lane & 15);
    int  hi = (lane >> 4) * 8;
    int  kb = ((v < 4) ? (v * 2) : (16 + (v - 4) * 2)) + hi;
    uint32_t word = 0;
#pragma unroll
    for (int e = 0; e < 2; ++e) {
        int K = kc * 32 + kb + e;
        uint16_t hb = 0;
        if (K < Ktot) {
            int k9 = K / Cin;
            int c  = K - k9 * Cin;
            hb = f32_to_bf16(w[((long)n * Cin + c) * 9 + k9]);
        }
        word |= ((uint32_t)hb) << (e * 16);
    }
    wsw[idx] = word;
}

// ---------------------------------------------------------------------------
// Layout converts
// ---------------------------------------------------------------------------
__global__ void nchw_f32_to_nhwc_bf16(const float* __restrict__ in,
                                      uint16_t* __restrict__ out,
                                      int C, int HW, long total) {
    long idx = (long)blockIdx.x * 256 + threadIdx.x;
    if (idx >= total) return;
    long CHW = (long)C * HW;
    int  b = (int)(idx / CHW);
    long r = idx - (long)b * CHW;
    int  c = (int)(r / HW);
    int  p = (int)(r - (long)c * HW);
    out[((long)b * HW + p) * C + c] = f32_to_bf16(in[idx]);
}

__global__ void nhwc_bf16_to_nchw_f32(const uint16_t* __restrict__ in,
                                      float* __restrict__ out,
                                      int C, int HW, long total) {
    long idx = (long)blockIdx.x * 256 + threadIdx.x;
    if (idx >= total) return;
    long CHW = (long)C * HW;
    int  b = (int)(idx / CHW);
    long r = idx - (long)b * CHW;
    int  c = (int)(r / HW);
    int  p = (int)(r - (long)c * HW);
    out[idx] = bf16_to_f32(in[((long)b * HW + p) * C + c]);
}

// ---------------------------------------------------------------------------
// 2x2 max-pool, NHWC bf16 (bf16 values exact -> max then re-encode is exact)
// ---------------------------------------------------------------------------
__global__ void maxpool2x2_nhwc(const uint16_t* __restrict__ in,
                                uint16_t* __restrict__ out,
                                int H, int W, int C, long total) {
    long idx = (long)blockIdx.x * 256 + threadIdx.x;
    if (idx >= total) return;
    int Wo = W >> 1, Ho = H >> 1;
    int c  = (int)(idx % C);
    long t = idx / C;
    int xo = (int)(t % Wo); t /= Wo;
    int yo = (int)(t % Ho);
    int b  = (int)(t / Ho);
    const uint16_t* p0 = in + (((long)b * H + yo * 2) * W + xo * 2) * C + c;
    float m = bf16_to_f32(p0[0]);
    m = fmaxf(m, bf16_to_f32(p0[C]));
    m = fmaxf(m, bf16_to_f32(p0[(long)W * C]));
    m = fmaxf(m, bf16_to_f32(p0[(long)W * C + C]));
    out[idx] = f32_to_bf16(m);
}

// ---------------------------------------------------------------------------
// Implicit-GEMM 3x3 SAME conv via V_WMMA_F32_16X16X32_BF16.
//  in  : NHWC bf16  [2, H, W, Cin]
//  wsw : swizzled weights (see repack_weights)
//  bias: f32 [Cout]
//  out : NHWC bf16  [2, H, W, Cout]
// Block = 128 threads (4 waves). Block tile = 128 pixels x 64 Cout.
// Wave w owns pixel rows [32w, 32w+32) (two 16-row A fragments) x 64 Cout
// -> 8 WMMA accumulators, each B fragment reused for two WMMAs.
// ---------------------------------------------------------------------------
#define TM 128
#define TK 32
#define A_STRIDE 36   // halves; padded so A-fragment reads are conflict-free

__global__ __launch_bounds__(128) void conv3x3_wmma_bf16(
    const uint16_t* __restrict__ in,
    const uint32_t* __restrict__ wsw,
    const float*    __restrict__ bias,
    uint16_t* __restrict__ out,
    int H, int W, int Cin, int Cout, int relu)
{
    const int Ktot = 9 * Cin;
    const int Kc   = (Ktot + TK - 1) / TK;
    const int HW   = H * W;
    const int NT   = Cout >> 4;
    const int tid  = threadIdx.x;
    const int lane = tid & 31;
    const int wave = tid >> 5;

    const int m0     = blockIdx.x * TM;
    const int ntile0 = blockIdx.y * 4;

    __shared__ __attribute__((aligned(16))) uint16_t lds_a[TM * A_STRIDE];
    __shared__ __attribute__((aligned(16))) uint32_t lds_b[1024];
    __shared__ int tap_off[TM * 9];

    // ---- once per block: per-pixel, per-tap base offsets (element units) ----
    for (int i = tid; i < TM * 9; i += 128) {
        int mi = i / 9;
        int t  = i - mi * 9;
        int p  = m0 + mi;
        int b  = p / HW;
        int rp = p - b * HW;
        int y  = rp / W;
        int x  = rp - y * W;
        int yy = y + t / 3 - 1;
        int xx = x + (t - (t / 3) * 3) - 1;
        tap_off[i] = (yy >= 0 && yy < H && xx >= 0 && xx < W)
                   ? ((b * H + yy) * W + xx) * Cin : -1;
    }

    v8f acc[8];
#pragma unroll
    for (int i = 0; i < 8; ++i) acc[i] = (v8f)(0.0f);

    // Per-lane fragment addressing (ISA 7.12.2, 16-bit layouts)
    const int frag_m0 = wave * 32 + (lane & 15);
    const int frag_hi = (lane >> 4) * 8;
    const bool fast   = (Cin & 31) == 0;   // true for all layers except layer 1

    for (int kc = 0; kc < Kc; ++kc) {
        __syncthreads();   // covers tap_off (kc==0) and prior-iter LDS reads

        // ---- stage A tile (im2col gather): 128 pixels x 32 K ----
        if (fast) {
            // 32-wide chunk never straddles a tap: k9 scalar, c = cb (+lane)
            int k9 = (kc * TK) / Cin;
            int cb = kc * TK - k9 * Cin + lane;
#pragma unroll 8
            for (int j = 0; j < TM / 4; ++j) {
                int mi  = wave + j * 4;
                int off = tap_off[mi * 9 + k9];            // uniform per wave
                uint16_t v = (off >= 0) ? in[off + cb] : (uint16_t)0;
                lds_a[mi * A_STRIDE + lane] = v;
            }
        } else {
            for (int i = tid; i < TM * TK; i += 128) {
                int mi = i >> 5;
                int kk = i & 31;
                int K  = kc * TK + kk;
                uint16_t v = 0;
                if (K < Ktot) {
                    int k9  = K / Cin;
                    int c   = K - k9 * Cin;
                    int off = tap_off[mi * 9 + k9];
                    if (off >= 0) v = in[off + c];
                }
                lds_a[mi * A_STRIDE + kk] = v;
            }
        }

        // ---- stage B tile: 4 n-tiles x 8 vgpr x 32 lanes of u32 ----
        {
            const uint32_t* src = wsw + ((long)kc * NT + ntile0) * 256;
#if defined(HAVE_ASYNC_B32)
#pragma unroll
            for (int j = 0; j < 8; ++j)
                async_copy_b32(src + tid + j * 128, &lds_b[tid + j * 128]);
            wait_async_lds();
#else
            for (int i = tid; i < 1024; i += 128) lds_b[i] = src[i];
#endif
            if (kc + 1 < Kc) {  // pull next chunk's weights toward L2/L0
                const uint32_t* nxt = wsw + ((long)(kc + 1) * NT + ntile0) * 256;
                __builtin_prefetch(nxt + tid, 0, 1);
            }
        }
        __syncthreads();

        // ---- load two A fragments (16x32 bf16 each) ----
        Frag16 a0, a1;
#pragma unroll
        for (int v = 0; v < 8; ++v) {
            int k = ((v < 4) ? (v * 2) : (16 + (v - 4) * 2)) + frag_hi;
            a0.u[v] = *(const uint32_t*)&lds_a[(frag_m0     ) * A_STRIDE + k];
            a1.u[v] = *(const uint32_t*)&lds_a[(frag_m0 + 16) * A_STRIDE + k];
        }

        // ---- 8 WMMAs: each B fragment reused for two A fragments ----
#pragma unroll
        for (int nt = 0; nt < 4; ++nt) {
            Frag16 b;
#pragma unroll
            for (int v = 0; v < 8; ++v) b.u[v] = lds_b[(nt * 8 + v) * 32 + lane];
            acc[nt]     = __builtin_amdgcn_wmma_f32_16x16x32_bf16(
                false, a0.bf, false, b.bf, (short)0, acc[nt],     false, false);
            acc[4 + nt] = __builtin_amdgcn_wmma_f32_16x16x32_bf16(
                false, a1.bf, false, b.bf, (short)0, acc[4 + nt], false, false);
        }
    }

    // ---- epilogue: bias + ReLU, store NHWC bf16 ----
    const int ncol  = lane & 15;
    const int rbase = (lane >> 4) * 8;
#pragma unroll
    for (int half = 0; half < 2; ++half) {
#pragma unroll
        for (int nt = 0; nt < 4; ++nt) {
            int   n  = blockIdx.y * 64 + nt * 16 + ncol;
            float bz = bias[n];
#pragma unroll
            for (int r = 0; r < 8; ++r) {
                int   m = m0 + wave * 32 + half * 16 + rbase + r;
                float v = acc[half * 4 + nt][r] + bz;
                if (relu) v = fmaxf(v, 0.0f);
                out[(long)m * Cout + n] = f32_to_bf16(v);
            }
        }
    }
}

// ---------------------------------------------------------------------------
// Host orchestration
// ---------------------------------------------------------------------------
extern "C" void kernel_launch(void* const* d_in, const int* in_sizes, int n_in,
                              void* d_out, int out_size, void* d_ws, size_t ws_size,
                              hipStream_t stream) {
    (void)in_sizes; (void)n_in; (void)out_size; (void)ws_size;

    const float* batch = (const float*)d_in[0];
    // d_in[1] = pooling_mask (unused by the reference)
    const float* Wt[11];
    const float* Bi[11];
    for (int i = 1; i <= 10; ++i) {
        Wt[i] = (const float*)d_in[2 + 2 * (i - 1)];
        Bi[i] = (const float*)d_in[3 + 2 * (i - 1)];
    }

    struct { int Cin, Cout; } cfg[11] = {
        {0,0},{3,64},{64,64},{64,128},{128,128},{128,256},
        {256,256},{256,256},{256,512},{512,512},{512,512}};

    // Workspace carving (all offsets 256B aligned)
    char*  ws  = (char*)d_ws;
    size_t off = 0;
    auto carve = [&](size_t bytes) -> void* {
        void* p = ws + off;
        off = (off + bytes + 255) & ~(size_t)255;
        return p;
    };
    const size_t ACT_BYTES = (size_t)2 * 384 * 384 * 64 * 2; // largest act, bf16
    uint16_t* actA = (uint16_t*)carve(ACT_BYTES);
    uint16_t* actB = (uint16_t*)carve(ACT_BYTES);

    uint32_t* wsw[11];
    long      wswWords[11];
    for (int i = 1; i <= 10; ++i) {
        int  Kc = (9 * cfg[i].Cin + 31) / 32;
        long words = (long)Kc * (cfg[i].Cout / 16) * 256;
        wswWords[i] = words;
        wsw[i] = (uint32_t*)carve((size_t)words * 4);
    }

    // 1) repack all weights into WMMA B-fragment layout (bf16)
    for (int i = 1; i <= 10; ++i) {
        long words  = wswWords[i];
        int  blocks = (int)((words + 255) / 256);
        repack_weights<<<blocks, 256, 0, stream>>>(Wt[i], wsw[i],
                                                   cfg[i].Cin, cfg[i].Cout, words);
    }

    // 2) input NCHW f32 -> NHWC bf16
    {
        long total = (long)2 * 3 * 384 * 384;
        nchw_f32_to_nhwc_bf16<<<(int)((total + 255) / 256), 256, 0, stream>>>(
            batch, actA, 3, 384 * 384, total);
    }

    auto conv = [&](const uint16_t* in, uint16_t* out, int layer, int H, int Wd) {
        int  Cin = cfg[layer].Cin, Cout = cfg[layer].Cout;
        long M = (long)2 * H * Wd;
        dim3 grid((unsigned)(M / TM), (unsigned)(Cout / 64));
        conv3x3_wmma_bf16<<<grid, 128, 0, stream>>>(in, wsw[layer], Bi[layer],
                                                    out, H, Wd, Cin, Cout, 1);
    };
    auto pool = [&](const uint16_t* in, uint16_t* out, int H, int Wd, int C) {
        long total = (long)2 * (H / 2) * (Wd / 2) * C;
        maxpool2x2_nhwc<<<(int)((total + 255) / 256), 256, 0, stream>>>(
            in, out, H, Wd, C, total);
    };

    // 3) layer sequence (ping-pong actA/actB)
    conv(actA, actB, 1, 384, 384);          // 3   -> 64  @384
    conv(actB, actA, 2, 384, 384);          // 64  -> 64  @384
    pool(actA, actB, 384, 384, 64);         // -> 192, C=64
    conv(actB, actA, 3, 192, 192);          // 64  -> 128 @192
    conv(actA, actB, 4, 192, 192);          // 128 -> 128 @192
    pool(actB, actA, 192, 192, 128);        // -> 96, C=128
    conv(actA, actB, 5, 96, 96);            // 128 -> 256 @96
    conv(actB, actA, 6, 96, 96);            // 256 -> 256 @96
    conv(actA, actB, 7, 96, 96);            // 256 -> 256 @96
    pool(actB, actA, 96, 96, 256);          // -> 48, C=256
    conv(actA, actB, 8, 48, 48);            // 256 -> 512 @48
    conv(actB, actA, 9, 48, 48);            // 512 -> 512 @48
    conv(actA, actB, 10, 48, 48);           // 512 -> 512 @48

    // 4) NHWC bf16 -> NCHW f32 into d_out
    {
        long total = (long)2 * 512 * 48 * 48;
        nhwc_bf16_to_nchw_f32<<<(int)((total + 255) / 256), 256, 0, stream>>>(
            actB, (float*)d_out, 512, 48 * 48, total);
    }
}
